// MultiHeadAttention_25074019074189
// MI455X (gfx1250) — matrix operand order repaired and checked by reading, not compiled, not run on hardware
//
#include <hip/hip_runtime.h>
#include <hip/hip_bf16.h>

// ---------------------------------------------------------------------------
// MultiHeadAttention forward for MI455X (gfx1250, wave32, WMMA f16->f32).
// B=2, T=2048, C=1024, H=16, HD=64.
//
// Pipeline:
//   0) f32_to_f16_kernel : one-pass convert of x / W_qkv / W_proj to f16 in ws
//                          (removes redundant per-tile cvt work from GEMM loops;
//                          halves inner-loop bytes; all operands L2-resident).
//   1) qkv_rope_kernel   : qkv = x @ Wqkv^T (f16 WMMA, f32 acc), RoPE on q/k,
//                          scatter Q[b,h,t,d], K[b,h,t,d], Vt[b,h,d,t] as f16.
//   2) flash_attn_kernel : per-wave 16-query flash attention, online softmax;
//                          S=Q K^T and O+=P V via v_wmma_f32_16x16x32_f16.
//   3) proj_kernel       : out = attn @ Wproj^T + bias (f16 WMMA, f32 out).
//
// WMMA operand layouts assumed (CDNA5 ISA 7.12.2, wave32):
//   A (16x32 f16): lane L: row m=L&15, hi=L>>4; half i -> k = 16*(i>>3)+8*hi+(i&7)
//   B (32x16 f16): lane L: col n=L&15, hi=L>>4; half i -> k = 16*(i>>3)+8*hi+(i&7)
//   C/D (16x16 f32): lane L: col n=L&15; VGPR r -> row m = r + 8*(L>>4)
// ---------------------------------------------------------------------------

typedef _Float16 half_t;
typedef __attribute__((ext_vector_type(16))) _Float16 v16h;
typedef __attribute__((ext_vector_type(8)))  _Float16 v8h;
typedef __attribute__((ext_vector_type(4)))  _Float16 v4h;
typedef __attribute__((ext_vector_type(8)))  float    v8f;
typedef __attribute__((ext_vector_type(4)))  float    v4f;

#define B_   2
#define T_   2048
#define C_   1024
#define H_   16
#define HD_  64
#define BT_  (B_ * T_)
#define C3_  (3 * C_)

__device__ inline v8f wmma_f16f32(v16h a, v16h b, v8f c) {
  return __builtin_amdgcn_wmma_f32_16x16x32_f16(false, a, false, b, (short)0, c,
                                                false, false);
}

// B-fragment: k = 8*hi + 0..7 and k = 16 + 8*hi + 0..7 along memory -> v16h
__device__ inline v16h load_bfrag_f16(const half_t* base, int hi) {
  const v8h* p0 = (const v8h*)(base + 8 * hi);
  const v8h* p1 = (const v8h*)(base + 16 + 8 * hi);
  v8h x0 = *p0, x1 = *p1;
  v16h r;
#pragma unroll
  for (int e = 0; e < 8; ++e) { r[e] = x0[e]; r[8 + e] = x1[e]; }
  return r;
}

// A-fragment from a row of 32 f16 values (row = this lane's matrix row).
__device__ inline v16h load_afrag_f16(const half_t* row, int hi) {
  const v8h* p0 = (const v8h*)(row + 8 * hi);        // k = 8*hi .. +7
  const v8h* p1 = (const v8h*)(row + 16 + 8 * hi);   // k = 16+8*hi .. +7
  v8h a = *p0, b = *p1;
  v16h r;
#pragma unroll
  for (int e = 0; e < 8; ++e) { r[e] = a[e]; r[8 + e] = b[e]; }
  return r;
}

// ---------------------------------------------------------------------------
// Kernel 0: elementwise f32 -> f16 (4 elements/thread, b128 in / b64 out).
// ---------------------------------------------------------------------------
__global__ __launch_bounds__(256) void f32_to_f16_kernel(
    const float* __restrict__ in, half_t* __restrict__ out, int n4) {
  int idx = blockIdx.x * 256 + threadIdx.x;
  if (idx < n4) {
    v4f f = *(const v4f*)(in + (size_t)idx * 4);
    v4h h;
#pragma unroll
    for (int e = 0; e < 4; ++e) h[e] = (half_t)f[e];
    *(v4h*)(out + (size_t)idx * 4) = h;
  }
}

// ---------------------------------------------------------------------------
// Kernel 1: qkv = x @ Wqkv^T, RoPE, scatter to Q/K/Vt (f16).
// grid = (BT/16, 3C/256), block = 128 (4 waves); each wave: 16(M) x 64(N).
// ---------------------------------------------------------------------------
__global__ __launch_bounds__(128) void qkv_rope_kernel(
    const half_t* __restrict__ Xh, const half_t* __restrict__ Wqkvh,
    const float* __restrict__ fcos, const float* __restrict__ fsin,
    half_t* __restrict__ Qh, half_t* __restrict__ Kh, half_t* __restrict__ Vt) {
  const int lane = threadIdx.x & 31;
  const int wave = threadIdx.x >> 5;
  const int nlo  = lane & 15;
  const int hi   = lane >> 4;
  const int m0   = blockIdx.x * 16;                 // token-tile base
  const int j0w  = blockIdx.y * 256 + wave * 64;    // output-col base for wave

  v8f acc[4];
#pragma unroll
  for (int u = 0; u < 4; ++u)
#pragma unroll
    for (int r = 0; r < 8; ++r) acc[u][r] = 0.f;

  const half_t* arow = Xh + (size_t)(m0 + nlo) * C_;
  for (int k0 = 0; k0 < C_; k0 += 32) {
    v16h a = load_afrag_f16(arow + k0, hi);
#pragma unroll
    for (int u = 0; u < 4; ++u) {
      const half_t* brow = Wqkvh + (size_t)(j0w + u * 16 + nlo) * C_ + k0;
      acc[u] = wmma_f16f32(a, load_bfrag_f16(brow, hi), acc[u]);
    }
  }

  // Epilogue: RoPE (q,k) via lane-pair shuffle; scatter to Q/K/Vt.
#pragma unroll
  for (int u = 0; u < 4; ++u) {
    const int jbase = j0w + u * 16;
    const int which = jbase >> 10;          // 0=q, 1=k, 2=v (tile never straddles)
    const int cc    = (jbase & (C_ - 1)) + nlo;
    const int h     = cc >> 6;
    const int d     = cc & (HD_ - 1);
#pragma unroll
    for (int r = 0; r < 8; ++r) {
      const int bt = m0 + r + 8 * hi;
      const int b  = bt >> 11;              // /T_
      const int t  = bt & (T_ - 1);
      float val = acc[u][r];
      if (which == 2) {
        // V stored transposed: Vt[b,h,d,t]
        Vt[(size_t)((b * H_ + h) * HD_ + d) * T_ + t] = (half_t)val;
      } else {
        float partner = __shfl_xor(val, 1, 32);   // (re,im) in adjacent lanes
        const int i = d >> 1;
        float c  = fcos[t * (HD_ / 2) + i];
        float sn = fsin[t * (HD_ / 2) + i];
        // even col: out_re = re*c - im*s ; odd col: out_im = re*s + im*c
        float outv = (lane & 1) ? (partner * sn + val * c)
                                : (val * c - partner * sn);
        half_t* dst = (which == 0) ? Qh : Kh;
        dst[(size_t)((b * H_ + h) * T_ + t) * HD_ + d] = (half_t)outv;
      }
    }
  }
}

// ---------------------------------------------------------------------------
// Kernel 2: flash attention. One wave = one 16-query tile of one (b,h).
// grid = B*H*(T/16)/4, block = 128 (4 waves). LDS: 1KB/wave for P transpose.
// ---------------------------------------------------------------------------
__global__ __launch_bounds__(128) void flash_attn_kernel(
    const half_t* __restrict__ Qh, const half_t* __restrict__ Kh,
    const half_t* __restrict__ Vt, half_t* __restrict__ AO) {
  __shared__ __align__(16) half_t Plds[4][16 * 32];

  const int lane = threadIdx.x & 31;
  const int wave = threadIdx.x >> 5;
  const int nlo  = lane & 15;
  const int hi   = lane >> 4;

  const int qt = blockIdx.x * 4 + wave;   // global q-tile id
  const int bh = qt >> 7;                 // T/16 = 128 q-tiles per (b,h)
  const int q0 = (qt & 127) * 16;
  const int b  = bh >> 4;
  const int h  = bh & 15;

  const half_t* Qp = Qh + (size_t)bh * T_ * HD_;
  const half_t* Kp = Kh + (size_t)bh * T_ * HD_;
  const half_t* Vp = Vt + (size_t)bh * HD_ * T_;
  half_t* myP = &Plds[wave][0];

  // Q fragments (rows = queries via lanes), pre-scaled by HD^-0.5 = 1/8 (exact).
  v16h qa[2];
#pragma unroll
  for (int g2 = 0; g2 < 2; ++g2) {
    qa[g2] = load_afrag_f16(Qp + (size_t)(q0 + nlo) * HD_ + 32 * g2, hi);
#pragma unroll
    for (int i = 0; i < 16; ++i) qa[g2][i] = qa[g2][i] * (half_t)0.125f;
  }

  float mrow[8], lrow[8];
  v8f o[4];
#pragma unroll
  for (int r = 0; r < 8; ++r) { mrow[r] = -1e30f; lrow[r] = 0.f; }
#pragma unroll
  for (int dt = 0; dt < 4; ++dt)
#pragma unroll
    for (int r = 0; r < 8; ++r) o[dt][r] = 0.f;

  for (int k0 = 0; k0 <= q0 + 15; k0 += 32) {   // causal: keys <= q0+15
    // ---- S = (Q/8) K^T for 32 keys (two 16-key column tiles) ----
    v8f s[2];
#pragma unroll
    for (int ct = 0; ct < 2; ++ct) {
      const int key = k0 + 16 * ct + nlo;       // this lane's key column
      v8f sc;
#pragma unroll
      for (int r = 0; r < 8; ++r) sc[r] = 0.f;
      sc = wmma_f16f32(qa[0], load_bfrag_f16(Kp + (size_t)key * HD_, hi), sc);
      sc = wmma_f16f32(qa[1], load_bfrag_f16(Kp + (size_t)key * HD_ + 32, hi), sc);
#pragma unroll
      for (int r = 0; r < 8; ++r) {             // causal mask
        const int q = q0 + r + 8 * hi;
        if (key > q) sc[r] = -1e30f;
      }
      s[ct] = sc;
    }

    // ---- online softmax (row = 16 lanes within each half-wave) ----
    float p0a[8], p1a[8];
#pragma unroll
    for (int r = 0; r < 8; ++r) {
      float mx = fmaxf(s[0][r], s[1][r]);
#pragma unroll
      for (int off = 1; off < 16; off <<= 1) mx = fmaxf(mx, __shfl_xor(mx, off, 32));
      const float mnew  = fmaxf(mrow[r], mx);
      const float alpha = __expf(mrow[r] - mnew);
      mrow[r] = mnew;
      float pa = __expf(s[0][r] - mnew);
      float pb = __expf(s[1][r] - mnew);
      float rs = pa + pb;
#pragma unroll
      for (int off = 1; off < 16; off <<= 1) rs += __shfl_xor(rs, off, 32);
      lrow[r] = lrow[r] * alpha + rs;
#pragma unroll
      for (int dt = 0; dt < 4; ++dt) o[dt][r] *= alpha;
      p0a[r] = pa; p1a[r] = pb;
    }

    // ---- P: C/D layout -> LDS -> A layout ----
#pragma unroll
    for (int r = 0; r < 8; ++r) {
      myP[(r + 8 * hi) * 32 + nlo]      = (half_t)p0a[r];
      myP[(r + 8 * hi) * 32 + 16 + nlo] = (half_t)p1a[r];
    }
    asm volatile("s_wait_dscnt 0" ::: "memory");   // cross-lane LDS visibility
    v16h pfrag = load_afrag_f16(myP + nlo * 32, hi);

    // ---- O += P V  (V stored [d][t] so B columns are contiguous) ----
#pragma unroll
    for (int dt = 0; dt < 4; ++dt) {
      const int d = 16 * dt + nlo;
      v16h vb = load_bfrag_f16(Vp + (size_t)d * T_ + k0, hi);
      o[dt] = wmma_f16f32(pfrag, vb, o[dt]);
    }
    asm volatile("s_wait_dscnt 0" ::: "memory");   // P reads done before rewrite
  }

  // ---- epilogue: O /= l, write attn output [b,t, h*64+d] as f16 ----
#pragma unroll
  for (int dt = 0; dt < 4; ++dt)
#pragma unroll
    for (int r = 0; r < 8; ++r) {
      const int t = q0 + r + 8 * hi;
      float outv = o[dt][r] / lrow[r];
      AO[(size_t)(b * T_ + t) * C_ + h * HD_ + 16 * dt + nlo] = (half_t)outv;
    }
}

// ---------------------------------------------------------------------------
// Kernel 3: out = attn @ Wproj^T + bias  (f32 output).
// grid = (BT/16, C/256), block = 128 (4 waves); each wave 16x64.
// ---------------------------------------------------------------------------
__global__ __launch_bounds__(128) void proj_kernel(
    const half_t* __restrict__ AO, const half_t* __restrict__ Wprojh,
    const float* __restrict__ bias, float* __restrict__ out) {
  const int lane = threadIdx.x & 31;
  const int wave = threadIdx.x >> 5;
  const int nlo  = lane & 15;
  const int hi   = lane >> 4;
  const int m0   = blockIdx.x * 16;
  const int j0w  = blockIdx.y * 256 + wave * 64;

  v8f acc[4];
#pragma unroll
  for (int u = 0; u < 4; ++u)
#pragma unroll
    for (int r = 0; r < 8; ++r) acc[u][r] = 0.f;

  const half_t* arow = AO + (size_t)(m0 + nlo) * C_;
  for (int k0 = 0; k0 < C_; k0 += 32) {
    v16h a = load_afrag_f16(arow + k0, hi);
#pragma unroll
    for (int u = 0; u < 4; ++u) {
      const half_t* brow = Wprojh + (size_t)(j0w + u * 16 + nlo) * C_ + k0;
      acc[u] = wmma_f16f32(a, load_bfrag_f16(brow, hi), acc[u]);
    }
  }

#pragma unroll
  for (int u = 0; u < 4; ++u) {
    const int n = j0w + u * 16 + nlo;
    const float bv = bias[n];
#pragma unroll
    for (int r = 0; r < 8; ++r) {
      const int bt = m0 + r + 8 * hi;
      out[(size_t)bt * C_ + n] = acc[u][r] + bv;
    }
  }
}

// ---------------------------------------------------------------------------
extern "C" void kernel_launch(void* const* d_in, const int* in_sizes, int n_in,
                              void* d_out, int out_size, void* d_ws, size_t ws_size,
                              hipStream_t stream) {
  (void)in_sizes; (void)n_in; (void)out_size; (void)ws_size;

  const float* x     = (const float*)d_in[0];
  const float* fcos  = (const float*)d_in[1];
  const float* fsin  = (const float*)d_in[2];
  const float* Wqkv  = (const float*)d_in[3];
  const float* Wproj = (const float*)d_in[4];
  const float* bias  = (const float*)d_in[5];
  float* out = (float*)d_out;

  // Workspace (f16), 48 MB total (L2 is 192 MB):
  //   Qh, Kh (b,h,t,d); Vt (b,h,d,t); AO (b,t,c); Xh; Wqkvh; Wprojh.
  const size_t NQK = (size_t)B_ * H_ * T_ * HD_;   // 4,194,304 halfs
  half_t* Qh     = (half_t*)d_ws;
  half_t* Kh     = Qh + NQK;
  half_t* Vt     = Kh + NQK;
  half_t* AO     = Vt + NQK;
  half_t* Xh     = AO + NQK;                        // BT*C halfs
  half_t* Wqkvh  = Xh + (size_t)BT_ * C_;           // 3C*C halfs
  half_t* Wprojh = Wqkvh + (size_t)C3_ * C_;        // C*C halfs

  // 0) one-pass f32 -> f16 materialization
  const int nx4 = (BT_ * C_) / 4, nwq4 = (C3_ * C_) / 4, nwp4 = (C_ * C_) / 4;
  f32_to_f16_kernel<<<nx4 / 256, 256, 0, stream>>>(x, Xh, nx4);
  f32_to_f16_kernel<<<nwq4 / 256, 256, 0, stream>>>(Wqkv, Wqkvh, nwq4);
  f32_to_f16_kernel<<<nwp4 / 256, 256, 0, stream>>>(Wproj, Wprojh, nwp4);

  // 1) QKV + RoPE
  qkv_rope_kernel<<<dim3(BT_ / 16, C3_ / 256), 128, 0, stream>>>(
      Xh, Wqkvh, fcos, fsin, Qh, Kh, Vt);
  // 2) causal flash attention
  flash_attn_kernel<<<dim3((B_ * H_ * (T_ / 16)) / 4), 128, 0, stream>>>(
      Qh, Kh, Vt, AO);
  // 3) output projection + bias
  proj_kernel<<<dim3(BT_ / 16, C_ / 256), 128, 0, stream>>>(
      AO, Wprojh, bias, out);
}
